// FDSE_25847113187507
// MI455X (gfx1250) — matrix operand order repaired and checked by reading, not compiled, not run on hardware
//
#include <hip/hip_runtime.h>
#include <hip/hip_bf16.h>
#include <math.h>

// ---------------- constants (match reference) ----------------
#define D_MODEL  128
#define D_STATE  16
#define D_CONVK  4
#define D_INNER  256
#define D_FF     512
#define DT_RANK  8
#define XDB_PAD  48            // dt_rank + 2*d_state = 40, padded to 48 (3 WMMA tiles)
#define T_TOK    16384         // tokens per pass (both passes)

typedef __attribute__((ext_vector_type(16))) _Float16 v16h;
typedef __attribute__((ext_vector_type(8)))  float    v8f;

__device__ __forceinline__ float silu_f(float x){ return x / (1.0f + __expf(-x)); }

// ---------------- fp32 -> f16 converter (flat, with optional row padding) ----
__global__ void k_cvt_w(const float* __restrict__ src, _Float16* __restrict__ dst,
                        int N, int K, int Np){
    int i = blockIdx.x * 256 + threadIdx.x;
    if (i >= Np * K) return;
    int n = i / K;
    dst[i] = (n < N) ? (_Float16)src[i] : (_Float16)0.0f;
}

__global__ void k_cvt_act(const float* __restrict__ src, _Float16* __restrict__ dst, int n){
    int i = blockIdx.x * 256 + threadIdx.x;
    if (i < n) dst[i] = (_Float16)src[i];
}

// ---------------- WMMA GEMM: C(MxN) = A(MxK) * W(NxK)^T  [+bias][+gelu] ------
// one wave32 per 16x16 C tile; pure-f16 operand streams -> v_wmma_f32_16x16x32_f16
template<int FUSE, typename TOUT>   // FUSE: 0 none, 1 +bias, 2 +bias +exact GELU
__global__ __launch_bounds__(32)
void k_gemm_wmma(const _Float16* __restrict__ A, int lda,
                 const _Float16* __restrict__ W, int ldw,
                 TOUT* __restrict__ C, int ldc,
                 const float* __restrict__ bias, int K){
    const int lane = threadIdx.x;
    const int mr   = lane & 15;       // row (A) / col (W,C) within tile
    const int g    = lane >> 4;       // lane group
    const int m0   = blockIdx.y << 4;
    const int n0   = blockIdx.x << 4;

    const _Float16* arow = A + (size_t)(m0 + mr) * lda;
    const _Float16* wrow = W + (size_t)(n0 + mr) * ldw;

    v8f acc = {};
    for (int k0 = 0; k0 < K; k0 += 32) {
        v16h af, bf;
#pragma unroll
        for (int v = 0; v < 8; ++v) {
            // A fragment (16x32): v<4 -> K=8g+2v ; v>=4 -> K=16+8g+2(v-4)
            int ka = (v < 4) ? (g * 8 + 2 * v) : (16 + g * 8 + 2 * (v - 4));
            af[2 * v]     = arow[k0 + ka];
            af[2 * v + 1] = arow[k0 + ka + 1];
            // B fragment (32x16): same K mapping as A
            int kb = ka;
            bf[2 * v]     = wrow[k0 + kb];
            bf[2 * v + 1] = wrow[k0 + kb + 1];
        }
        acc = __builtin_amdgcn_wmma_f32_16x16x32_f16(
                  false, af, false, bf, (short)0, acc, false, false);
    }

    const int n = n0 + mr;
    float bv = (FUSE > 0) ? bias[n] : 0.0f;
#pragma unroll
    for (int r = 0; r < 8; ++r) {
        int   m = m0 + g * 8 + r;           // C layout: vgpr r -> row r + 8*g
        float v = acc[r] + bv;
        if (FUSE == 2) v = 0.5f * v * (1.0f + erff(v * 0.70710678118654752f));
        C[(size_t)m * ldc + n] = (TOUT)v;
    }
}

// ---------------- causal depthwise conv (k=4) + bias + SiLU ----------------
// writes fp32 (for scan) and f16 (for x_proj GEMM)
__global__ void k_conv_silu(const float* __restrict__ xz,   // T x 512 (x = cols 0..255)
                            const float* __restrict__ cw,   // 256 x 4
                            const float* __restrict__ cb,   // 256
                            float* __restrict__ xc,         // T x 256 fp32
                            _Float16* __restrict__ xch,     // T x 256 f16
                            int L){
    int idx = blockIdx.x * 256 + threadIdx.x;
    if (idx >= T_TOK * D_INNER) return;
    int c = idx & (D_INNER - 1);
    int t = idx >> 8;
    int l = t % L;
    float acc = cb[c];
#pragma unroll
    for (int j = 0; j < D_CONVK; ++j) {
        int li = l - (D_CONVK - 1) + j;
        if (li >= 0)
            acc += cw[c * D_CONVK + j] * xz[(size_t)(t - (D_CONVK - 1) + j) * (2 * D_INNER) + c];
    }
    float r = silu_f(acc);
    xc[idx]  = r;
    xch[idx] = (_Float16)r;
}

// ---------------- dt = softplus(xdb[:, :8] @ dt_w^T + dt_b) ----------------
__global__ void k_dt(const float* __restrict__ xdb, const float* __restrict__ dtw,
                     const float* __restrict__ dtb, float* __restrict__ dt){
    int idx = blockIdx.x * 256 + threadIdx.x;
    if (idx >= T_TOK * D_INNER) return;
    int c = idx & (D_INNER - 1);
    int t = idx >> 8;
    const float* row = xdb + (size_t)t * XDB_PAD;
    float acc = dtb[c];
#pragma unroll
    for (int r = 0; r < DT_RANK; ++r) acc += row[r] * dtw[c * DT_RANK + r];
    dt[idx] = (acc > 20.0f) ? acc : log1pf(__expf(acc));
}

// ---------------- selective scan: h = exp(dt*A)*h + dt*B*x ; y = <h,C> ------
// one block per sequence, one thread per channel, h[16] in registers;
// gated output written directly as f16 for the out_proj WMMA
__global__ __launch_bounds__(D_INNER)
void k_scan(const float* __restrict__ xdb,   // T x 48  (dt|B|C)
            const float* __restrict__ dt,    // T x 256
            const float* __restrict__ xc,    // T x 256
            const float* __restrict__ xz,    // T x 512 (z = cols 256..511)
            const float* __restrict__ A_log, // 256 x 16
            const float* __restrict__ Dp,    // 256
            _Float16* __restrict__ ysh,      // T x 256 f16
            int L){
    int n = blockIdx.x;
    int c = threadIdx.x;
    __shared__ float sB[D_STATE], sC[D_STATE];
    float A[D_STATE], h[D_STATE];
#pragma unroll
    for (int j = 0; j < D_STATE; ++j) { A[j] = -__expf(A_log[c * D_STATE + j]); h[j] = 0.0f; }
    float Dc = Dp[c];
    for (int l = 0; l < L; ++l) {
        size_t t = (size_t)n * L + l;
        if (c < D_STATE)            sB[c]            = xdb[t * XDB_PAD + DT_RANK + c];
        else if (c < 2 * D_STATE)   sC[c - D_STATE]  = xdb[t * XDB_PAD + DT_RANK + D_STATE + (c - D_STATE)];
        __syncthreads();
        float dtv = dt[t * D_INNER + c];
        float xv  = xc[t * D_INNER + c];
        float y   = 0.0f;
#pragma unroll
        for (int j = 0; j < D_STATE; ++j) {
            float dA = __expf(dtv * A[j]);
            h[j] = dA * h[j] + dtv * sB[j] * xv;
            y   += h[j] * sC[j];
        }
        y += xv * Dc;
        float zv = xz[t * (2 * D_INNER) + D_INNER + c];
        ysh[t * D_INNER + c] = (_Float16)(y * silu_f(zv));
        __syncthreads();
    }
}

// ---------------- out = rmsnorm(a + b) * w ; dual-type / transposed store ---
// mode 0: linear (out fp32; out_h optional f16 mirror for next GEMM)
// mode 2: pass1 token (b*32+ts)*128+cyc -> pass2 order (b*128+cyc)*32+ts
// mode 1: pass2 token (b*128+cyc)*32+ts -> final layout [b, ts, cyc, d]
__global__ __launch_bounds__(D_MODEL)
void k_add_rmsnorm(const float* __restrict__ a, const float* __restrict__ b,
                   const float* __restrict__ w, float* __restrict__ out,
                   _Float16* __restrict__ out_h, int mode){
    int t = blockIdx.x;
    int i = threadIdx.x;
    __shared__ float red[D_MODEL];
    float v = a[(size_t)t * D_MODEL + i] + b[(size_t)t * D_MODEL + i];
    red[i] = v * v;
    __syncthreads();
    for (int s = D_MODEL / 2; s > 0; s >>= 1) {
        if (i < s) red[i] += red[i + s];
        __syncthreads();
    }
    float r = v * rsqrtf(red[0] * (1.0f / D_MODEL) + 1e-6f) * w[i];
    size_t o;
    if (mode == 0) {
        o = (size_t)t * D_MODEL + i;
    } else if (mode == 2) {
        int cyc = t & 127, rest = t >> 7, ts = rest & 31, bb = rest >> 5;
        o = ((((size_t)bb * 128 + cyc) * 32) + ts) * D_MODEL + i;
    } else {
        int ts = t & 31, rest = t >> 5, cyc = rest & 127, bb = rest >> 7;
        o = ((((size_t)bb * 32 + ts) * 128) + cyc) * D_MODEL + i;
    }
    out[o] = r;
    if (out_h) out_h[(size_t)t * D_MODEL + i] = (_Float16)r;
}

// ---------------- host-side pass orchestration ----------------
// params order: 0 in_proj 1 conv_w 2 conv_b 3 x_proj 4 dt_w 5 dt_b 6 A_log
//               7 D 8 out_proj 9 mlp_w1 10 mlp_b1 11 mlp_w2 12 mlp_b2 13 norm1_w
static void run_pass(const float* u, const float* const* P,
                     float* ws_f, _Float16* ws_h, _Float16* wh,
                     float* out, int L, int out_mode, hipStream_t stream){
    const size_t T = T_TOK;
    // fp32 intermediates
    float* xz  = ws_f;                        // T x 512
    float* xc  = ws_f + T * 512;              // T x 256
    float* xdb = ws_f + T * 768;              // T x 48
    float* dt  = ws_f + T * 816;              // T x 256
    float* ym  = ws_f + T * 1072;             // T x 128 (reused as h2)
    float* z1  = ws_f + T * 1200;             // T x 128
    // f16 activation mirrors
    _Float16* uh  = ws_h;                     // T x 128
    _Float16* xch = ws_h + T * 128;           // T x 256
    _Float16* ysh = ws_h + T * 384;           // T x 256
    _Float16* z1h = ws_h + T * 640;           // T x 128
    _Float16* h1h = ws_h + T * 768;           // T x 512
    // f16 weights
    _Float16* w_in  = wh;                     // 512 x 128
    _Float16* w_xp  = wh + 65536;             // 48  x 256 (rows 40..47 zero)
    _Float16* w_out = wh + 77824;             // 128 x 256
    _Float16* w_m1  = wh + 110592;            // 512 x 128
    _Float16* w_m2  = wh + 176128;            // 128 x 512

    // convert weights + input activations to f16
    k_cvt_w<<<(512 * 128 + 255) / 256, 256, 0, stream>>>(P[0],  w_in,  512, 128, 512);
    k_cvt_w<<<(48  * 256 + 255) / 256, 256, 0, stream>>>(P[3],  w_xp,  40,  256, 48);
    k_cvt_w<<<(128 * 256 + 255) / 256, 256, 0, stream>>>(P[8],  w_out, 128, 256, 128);
    k_cvt_w<<<(512 * 128 + 255) / 256, 256, 0, stream>>>(P[9],  w_m1,  512, 128, 512);
    k_cvt_w<<<(128 * 512 + 255) / 256, 256, 0, stream>>>(P[11], w_m2,  128, 512, 128);
    k_cvt_act<<<(T_TOK * 128 + 255) / 256, 256, 0, stream>>>(u, uh, T_TOK * 128);

    dim3 blk(32);
    // xz = u @ in_proj^T                       (T x 128) -> (T x 512)
    k_gemm_wmma<0, float><<<dim3(512 / 16, T_TOK / 16), blk, 0, stream>>>(
        uh, D_MODEL, w_in, 128, xz, 512, nullptr, 128);
    // causal depthwise conv + SiLU (fp32 + f16 outputs)
    k_conv_silu<<<T_TOK * D_INNER / 256, 256, 0, stream>>>(xz, P[1], P[2], xc, xch, L);
    // xdb = xc @ x_proj^T                      (T x 256) -> (T x 48)
    k_gemm_wmma<0, float><<<dim3(XDB_PAD / 16, T_TOK / 16), blk, 0, stream>>>(
        xch, D_INNER, w_xp, 256, xdb, XDB_PAD, nullptr, 256);
    // dt = softplus(.)
    k_dt<<<T_TOK * D_INNER / 256, 256, 0, stream>>>(xdb, P[4], P[5], dt);
    // selective scan (+ D skip + SiLU gate) -> f16
    k_scan<<<T_TOK / L, D_INNER, 0, stream>>>(xdb, dt, xc, xz, P[6], P[7], ysh, L);
    // ym = ys @ out_proj^T                     (T x 256) -> (T x 128)
    k_gemm_wmma<0, float><<<dim3(D_MODEL / 16, T_TOK / 16), blk, 0, stream>>>(
        ysh, D_INNER, w_out, 256, ym, D_MODEL, nullptr, 256);
    // z1 = rmsnorm(ym + u) * norm1_w  (fp32 + f16)
    k_add_rmsnorm<<<T_TOK, D_MODEL, 0, stream>>>(ym, u, P[13], z1, z1h, 0);
    // h1 = gelu(z1 @ w1^T + b1)                (T x 128) -> (T x 512) f16
    k_gemm_wmma<2, _Float16><<<dim3(D_FF / 16, T_TOK / 16), blk, 0, stream>>>(
        z1h, D_MODEL, w_m1, 128, h1h, D_FF, P[10], 128);
    // h2 = h1 @ w2^T + b2                      (T x 512) -> (T x 128), reuse ym
    k_gemm_wmma<1, float><<<dim3(D_MODEL / 16, T_TOK / 16), blk, 0, stream>>>(
        h1h, D_FF, w_m2, 512, ym, D_MODEL, P[12], 512);
    // out = rmsnorm(h2 + z1) * norm1_w  (with layout remap)
    k_add_rmsnorm<<<T_TOK, D_MODEL, 0, stream>>>(ym, z1, P[13], out, nullptr, out_mode);
}

extern "C" void kernel_launch(void* const* d_in, const int* in_sizes, int n_in,
                              void* d_out, int out_size, void* d_ws, size_t ws_size,
                              hipStream_t stream) {
    (void)in_sizes; (void)n_in; (void)out_size; (void)ws_size;
    const float* x = (const float*)d_in[0];
    const float* Pt[14]; const float* Pd[14];
    for (int i = 0; i < 14; ++i) Pt[i] = (const float*)d_in[1 + i];
    for (int i = 0; i < 14; ++i) Pd[i] = (const float*)d_in[15 + i];

    const size_t T = T_TOK;
    float*     ws_f = (float*)d_ws;                         // T*1328 fp32 intermediates
    float*     u2   = ws_f + T * 1328;                      // T x 128 (pass2 input)
    _Float16*  ws_h = (_Float16*)(ws_f + T * 1456);         // T*1280 f16 activations
    _Float16*  wh   = ws_h + T * 1280;                      // 241664 f16 weights

    // pass 1: 128 sequences of L=128 (scan over cycle axis), output in pass2 token order
    run_pass(x,  Pt, ws_f, ws_h, wh, u2,            128, /*mode=*/2, stream);
    // pass 2: 512 sequences of L=32 (scan over dim axis), output in final [b,ts,cyc,d]
    run_pass(u2, Pd, ws_f, ws_h, wh, (float*)d_out,  32, /*mode=*/1, stream);
}
